// ScopeGrammarLanguageModel_31181462569354
// MI455X (gfx1250) — matrix operand order repaired and checked by reading, not compiled, not run on hardware
//
#include <hip/hip_runtime.h>
#include <cstdint>
#include <cstddef>

// ============================================================================
// ScopeGrammarLanguageModel for MI455X (gfx1250), wave32 + WMMA bf16.
//
//  - All GEMMs use V_WMMA_F32_16X16X32_BF16 (fp32 WMMA on CDNA5 is only K=4;
//    bf16 gives 8x K-depth per issue, f32 accumulation).
//  - Recurrent-step GEMMs: 16x64 strip per wave (1 A frag -> 4 WMMAs),
//    K-loop software-pipelined (next K-block loads overlap current WMMAs,
//    verified: staggered s_wait_loadcnt 6/4/2/0 in round-2 asm).
//  - Output-head GEMMs (M=8192): 32x64 per wave (2 A frags x 4 B frags ->
//    8 WMMAs per 12 b128 loads) for higher matrix-op density; plenty of
//    waves remain (2048+) so occupancy is preserved.
//  - Weights (~25 MB bf16) are L2-resident (192 MB); no LDS staging needed.
//    Output store floor: 328 MB / 23.3 TB/s ~= 14 us -> matrix-core bound.
//  - Sequential scan -> per-step kernel pipeline; stack ping-pongs
//    (f32 master + bf16 operand copy) to avoid cross-block WAR races.
// ============================================================================

#define B_      32
#define S_      256
#define E_      512
#define H_      512
#define STACK_  32
#define VOCAB_  10000
#define KW_     1000
#define IDV_    (VOCAB_ - KW_)   // 9000
#define KWPAD_  1024             // pad to multiple of 64 for 4-wide N blocking
#define IDPAD_  9024

typedef __bf16 bf16;
typedef __attribute__((ext_vector_type(16))) __bf16 v16bf;
typedef __attribute__((ext_vector_type(8)))  float  v8f;

union F16x16 { v16bf v; uint4 q[2]; };

static __device__ __forceinline__ v8f wmma_bf16(v16bf a, v16bf b, v8f c) {
  // 8 args: (neg_a, A, neg_b, B, c_mod, C, reuse_a, reuse_b)
  return __builtin_amdgcn_wmma_f32_16x16x32_bf16(false, a, false, b,
                                                 (short)0, c, false, false);
}

// A fragment: 16x32 bf16, row-major source. Lane L (&15) = row M.
// lanes 0-15 hold K = k0+{0..7,16..23}; lanes 16-31 hold K = k0+{8..15,24..31}.
static __device__ __forceinline__ v16bf load_A(const bf16* __restrict__ p, int ld,
                                               int row, int k0, int lane) {
  const bf16* q = p + (size_t)row * ld + k0 + ((lane >> 4) << 3);
  F16x16 f;
  f.q[0] = *reinterpret_cast<const uint4*>(q);
  f.q[1] = *reinterpret_cast<const uint4*>(q + 16);
  return f.v;
}

// B fragment: B^T stored [N, K] row-major. Lane L (&15) = column N.
// lanes 0-15 hold K = k0+{0..7,16..23}; lanes 16-31 hold K = k0+{8..15,24..31}.
static __device__ __forceinline__ v16bf load_B(const bf16* __restrict__ p, int ld,
                                               int nrow, int k0, int lane) {
  const bf16* q = p + (size_t)nrow * ld + k0 + ((lane >> 4) << 3);
  F16x16 f;
  f.q[0] = *reinterpret_cast<const uint4*>(q);
  f.q[1] = *reinterpret_cast<const uint4*>(q + 16);
  return f.v;
}

// Split A for stack update: K<512 -> stack_bf16[row], else o_bf16[row>>5].
// K-blocks of 32 never straddle the 512 boundary.
static __device__ __forceinline__ v16bf load_A_split(const bf16* __restrict__ s0,
                                                     const bf16* __restrict__ s1,
                                                     int row, int k0, int lane) {
  int kk = k0 + ((lane >> 4) << 3);
  const bf16* q = (k0 < H_) ? (s0 + (size_t)row * H_ + kk)
                            : (s1 + (size_t)(row >> 5) * H_ + (kk - H_));
  F16x16 f;
  f.q[0] = *reinterpret_cast<const uint4*>(q);
  f.q[1] = *reinterpret_cast<const uint4*>(q + 16);
  return f.v;
}

// Fused hcat A: row b; K<512 -> stack[b*STACK + sidx[b]], else emb_t[b].
static __device__ __forceinline__ v16bf load_A_hcat(const bf16* __restrict__ stB,
                                                    const bf16* __restrict__ embt,
                                                    const int* __restrict__ sidxT,
                                                    int b, int k0, int lane) {
  int kk = k0 + ((lane >> 4) << 3);
  const bf16* q;
  if (k0 < H_)
    q = stB + ((size_t)(b * STACK_ + sidxT[(size_t)b * S_])) * H_ + kk;
  else
    q = embt + (size_t)b * S_ * E_ + (kk - H_);
  F16x16 f;
  f.q[0] = *reinterpret_cast<const uint4*>(q);
  f.q[1] = *reinterpret_cast<const uint4*>(q + 16);
  return f.v;
}

static __device__ __forceinline__ float sigm(float x) {
  return 1.0f / (1.0f + expf(-x));
}

#define ACC_ZERO {0.f, 0.f, 0.f, 0.f, 0.f, 0.f, 0.f, 0.f}

// Pipelined 16x64-per-wave GEMM body: shared A fragment, 4 B fragments,
// next K-block loaded before current WMMAs issue.
#define GEMM4_BODY(LOAD_A_CUR, LOAD_A_NXT)                                     \
  v8f acc0 = ACC_ZERO, acc1 = ACC_ZERO, acc2 = ACC_ZERO, acc3 = ACC_ZERO;      \
  v16bf a  = LOAD_A_CUR;                                                       \
  v16bf b0 = load_B(Bt, ldb, nb,      0, lane);                                \
  v16bf b1 = load_B(Bt, ldb, nb + 16, 0, lane);                                \
  v16bf b2 = load_B(Bt, ldb, nb + 32, 0, lane);                                \
  v16bf b3 = load_B(Bt, ldb, nb + 48, 0, lane);                                \
  for (int k0 = 32; k0 < K; k0 += 32) {                                        \
    v16bf an = LOAD_A_NXT;                                                     \
    v16bf c0 = load_B(Bt, ldb, nb,      k0, lane);                             \
    v16bf c1 = load_B(Bt, ldb, nb + 16, k0, lane);                             \
    v16bf c2 = load_B(Bt, ldb, nb + 32, k0, lane);                             \
    v16bf c3 = load_B(Bt, ldb, nb + 48, k0, lane);                             \
    acc0 = wmma_bf16(a, b0, acc0);                                             \
    acc1 = wmma_bf16(a, b1, acc1);                                             \
    acc2 = wmma_bf16(a, b2, acc2);                                             \
    acc3 = wmma_bf16(a, b3, acc3);                                             \
    a = an; b0 = c0; b1 = c1; b2 = c2; b3 = c3;                                \
  }                                                                            \
  acc0 = wmma_bf16(a, b0, acc0);                                               \
  acc1 = wmma_bf16(a, b1, acc1);                                               \
  acc2 = wmma_bf16(a, b2, acc2);                                               \
  acc3 = wmma_bf16(a, b3, acc3);

// Pipelined 32x64-per-wave GEMM body: 2 A fragments x 4 B fragments
// -> 8 WMMAs per K-block (12 b128 loads).
#define GEMM8_BODY(LA0_CUR, LA1_CUR, LA0_NXT, LA1_NXT)                         \
  v8f acc0 = ACC_ZERO, acc1 = ACC_ZERO, acc2 = ACC_ZERO, acc3 = ACC_ZERO;      \
  v8f acc4 = ACC_ZERO, acc5 = ACC_ZERO, acc6 = ACC_ZERO, acc7 = ACC_ZERO;      \
  v16bf a0 = LA0_CUR;                                                          \
  v16bf a1 = LA1_CUR;                                                          \
  v16bf b0 = load_B(Bt, ldb, nb,      0, lane);                                \
  v16bf b1 = load_B(Bt, ldb, nb + 16, 0, lane);                                \
  v16bf b2 = load_B(Bt, ldb, nb + 32, 0, lane);                                \
  v16bf b3 = load_B(Bt, ldb, nb + 48, 0, lane);                                \
  for (int k0 = 32; k0 < K; k0 += 32) {                                        \
    v16bf a0n = LA0_NXT;                                                       \
    v16bf a1n = LA1_NXT;                                                       \
    v16bf c0 = load_B(Bt, ldb, nb,      k0, lane);                             \
    v16bf c1 = load_B(Bt, ldb, nb + 16, k0, lane);                             \
    v16bf c2 = load_B(Bt, ldb, nb + 32, k0, lane);                             \
    v16bf c3 = load_B(Bt, ldb, nb + 48, k0, lane);                             \
    acc0 = wmma_bf16(a0, b0, acc0);                                            \
    acc1 = wmma_bf16(a0, b1, acc1);                                            \
    acc2 = wmma_bf16(a0, b2, acc2);                                            \
    acc3 = wmma_bf16(a0, b3, acc3);                                            \
    acc4 = wmma_bf16(a1, b0, acc4);                                            \
    acc5 = wmma_bf16(a1, b1, acc5);                                            \
    acc6 = wmma_bf16(a1, b2, acc6);                                            \
    acc7 = wmma_bf16(a1, b3, acc7);                                            \
    a0 = a0n; a1 = a1n; b0 = c0; b1 = c1; b2 = c2; b3 = c3;                    \
  }                                                                            \
  acc0 = wmma_bf16(a0, b0, acc0);                                              \
  acc1 = wmma_bf16(a0, b1, acc1);                                              \
  acc2 = wmma_bf16(a0, b2, acc2);                                              \
  acc3 = wmma_bf16(a0, b3, acc3);                                              \
  acc4 = wmma_bf16(a1, b0, acc4);                                              \
  acc5 = wmma_bf16(a1, b1, acc5);                                              \
  acc6 = wmma_bf16(a1, b2, acc6);                                              \
  acc7 = wmma_bf16(a1, b3, acc7);

// ---------------------------------------------------------------------------
// Head GEMM #1: C = relu(A[M,K] @ Bt[N,K]^T + bias) -> bf16. 32x64 per wave.
// M = 8192, K = 512, N = 512 (Ngrp = 8).
// ---------------------------------------------------------------------------
__global__ void k_gemm8(const bf16* __restrict__ A, int lda,
                        const bf16* __restrict__ Bt, int ldb,
                        const float* __restrict__ bias,
                        bf16* __restrict__ Cb, int ldc,
                        int M, int Ngrp, int K) {
  int wid  = (blockIdx.x * blockDim.x + threadIdx.x) >> 5;
  int lane = threadIdx.x & 31;
  int tm = wid / Ngrp, tg = wid - tm * Ngrp;
  if (tm * 32 >= M) return;
  int ar0 = tm * 32 + (lane & 15);
  int ar1 = ar0 + 16;
  int nb  = tg * 64 + (lane & 15);
  GEMM8_BODY(load_A(A, lda, ar0, 0, lane), load_A(A, lda, ar1, 0, lane),
             load_A(A, lda, ar0, k0, lane), load_A(A, lda, ar1, k0, lane))
  v8f accs[8] = {acc0, acc1, acc2, acc3, acc4, acc5, acc6, acc7};
  int m0 = tm * 32 + ((lane >> 4) << 3);
#pragma unroll
  for (int h = 0; h < 2; ++h) {
    int mb = m0 + h * 16;
#pragma unroll
    for (int j = 0; j < 4; ++j) {
      int n = tg * 64 + j * 16 + (lane & 15);
      float bz = bias[n];
#pragma unroll
      for (int r = 0; r < 8; ++r)
        Cb[(size_t)(mb + r) * ldc + n] = (bf16)fmaxf(accs[h * 4 + j][r] + bz, 0.f);
    }
  }
}

// ---------------------------------------------------------------------------
// Keyword head #2: out[:, 0:1000], terminal_mask -> -inf. 32x64 per wave.
// M = 8192, K = 512, Npad = 1024 (Ngrp = 16).
// ---------------------------------------------------------------------------
__global__ void k_gemm_kw8(const bf16* __restrict__ A, const bf16* __restrict__ Bt,
                           const float* __restrict__ bias,
                           const unsigned char* __restrict__ tmask,
                           float* __restrict__ out, int Ngrp) {
  int wid  = (blockIdx.x * blockDim.x + threadIdx.x) >> 5;
  int lane = threadIdx.x & 31;
  int tm = wid / Ngrp, tg = wid - tm * Ngrp;
  if (tm * 32 >= B_ * S_) return;
  int ar0 = tm * 32 + (lane & 15);
  int ar1 = ar0 + 16;
  int nb  = tg * 64 + (lane & 15);
  const int K = H_, ldb = H_;
  GEMM8_BODY(load_A(A, H_, ar0, 0, lane), load_A(A, H_, ar1, 0, lane),
             load_A(A, H_, ar0, k0, lane), load_A(A, H_, ar1, k0, lane))
  v8f accs[8] = {acc0, acc1, acc2, acc3, acc4, acc5, acc6, acc7};
  int m0 = tm * 32 + ((lane >> 4) << 3);
#pragma unroll
  for (int h = 0; h < 2; ++h) {
    int mb = m0 + h * 16;
#pragma unroll
    for (int j = 0; j < 4; ++j) {
      int n = tg * 64 + j * 16 + (lane & 15);
      if (n >= KW_) continue;
      float bz = bias[n];
#pragma unroll
      for (int r = 0; r < 8; ++r) {
        int m = mb + r;
        float v = accs[h * 4 + j][r] + bz;
        if (tmask[(size_t)m * KW_ + n]) v = -INFINITY;
        out[(size_t)m * VOCAB_ + n] = v;
      }
    }
  }
}

// ---------------------------------------------------------------------------
// Identifier head #2: out[:, 1000:10000], has_identifier -> -inf. 32x64/wave.
// M = 8192, K = 512, Npad = 9024 (Ngrp = 141).
// ---------------------------------------------------------------------------
__global__ void k_gemm_ident8(const bf16* __restrict__ A, const bf16* __restrict__ Bt,
                              const float* __restrict__ bias,
                              const unsigned char* __restrict__ hid,
                              float* __restrict__ out, int Ngrp) {
  int wid  = (blockIdx.x * blockDim.x + threadIdx.x) >> 5;
  int lane = threadIdx.x & 31;
  int tm = wid / Ngrp, tg = wid - tm * Ngrp;
  if (tm * 32 >= B_ * S_) return;
  int ar0 = tm * 32 + (lane & 15);
  int ar1 = ar0 + 16;
  int nb  = tg * 64 + (lane & 15);
  const int K = H_, ldb = H_;
  GEMM8_BODY(load_A(A, H_, ar0, 0, lane), load_A(A, H_, ar1, 0, lane),
             load_A(A, H_, ar0, k0, lane), load_A(A, H_, ar1, k0, lane))
  v8f accs[8] = {acc0, acc1, acc2, acc3, acc4, acc5, acc6, acc7};
  int m0 = tm * 32 + ((lane >> 4) << 3);
#pragma unroll
  for (int h = 0; h < 2; ++h) {
    int mb = m0 + h * 16;
#pragma unroll
    for (int j = 0; j < 4; ++j) {
      int n = tg * 64 + j * 16 + (lane & 15);
      if (n >= IDV_) continue;
      float bz = bias[n];
#pragma unroll
      for (int r = 0; r < 8; ++r) {
        int m = mb + r;
        float v = accs[h * 4 + j][r] + bz;
        if (hid[m]) v = -INFINITY;
        out[(size_t)m * VOCAB_ + KW_ + n] = v;
      }
    }
  }
}

// ---------------------------------------------------------------------------
// Step GEMM 1 with fused hcat gather:
//   tmp1 = relu([stack[b, sidx[b,t]] | emb_t] @ st_W1 + st_b1)   (bf16 out)
// M = 32, K = 1024, N = 512 (Ngrp = 8).
// ---------------------------------------------------------------------------
__global__ void k_gemm_hcat4(const bf16* __restrict__ stB,
                             const bf16* __restrict__ embt,       // + t*E_
                             const int* __restrict__ sidxT,       // + t
                             const bf16* __restrict__ Bt, int ldb,
                             const float* __restrict__ bias,
                             bf16* __restrict__ Cb, int Ngrp, int K) {
  int wid  = (blockIdx.x * blockDim.x + threadIdx.x) >> 5;
  int lane = threadIdx.x & 31;
  int tm = wid / Ngrp, tg = wid - tm * Ngrp;
  if (tm * 16 >= B_) return;
  int ar = tm * 16 + (lane & 15);
  int nb = tg * 64 + (lane & 15);
  GEMM4_BODY(load_A_hcat(stB, embt, sidxT, ar, 0, lane),
             load_A_hcat(stB, embt, sidxT, ar, k0, lane))
  v8f accs[4] = {acc0, acc1, acc2, acc3};
  int m0 = tm * 16 + ((lane >> 4) << 3);
#pragma unroll
  for (int j = 0; j < 4; ++j) {
    int n = tg * 64 + j * 16 + (lane & 15);
    float bz = bias[n];
#pragma unroll
    for (int r = 0; r < 8; ++r)
      Cb[(size_t)(m0 + r) * H_ + n] = (bf16)fmaxf(accs[j][r] + bz, 0.f);
  }
}

// ---------------------------------------------------------------------------
// scoped = tmp1 @ st_W2t + st_b2 ; x = is_identifier ? scoped : emb_t (bf16)
// M = 32, K = 512, N = 512 (Ngrp = 8).
// ---------------------------------------------------------------------------
__global__ void k_gemm_scoped4(const bf16* __restrict__ A,
                               const bf16* __restrict__ Bt, int ldb,
                               const float* __restrict__ bias,
                               const bf16* __restrict__ embt,           // + t*E_
                               const unsigned char* __restrict__ isidT, // + t
                               bf16* __restrict__ X, int Ngrp, int K) {
  int wid  = (blockIdx.x * blockDim.x + threadIdx.x) >> 5;
  int lane = threadIdx.x & 31;
  int tm = wid / Ngrp, tg = wid - tm * Ngrp;
  if (tm * 16 >= B_) return;
  int ar = tm * 16 + (lane & 15);
  int nb = tg * 64 + (lane & 15);
  GEMM4_BODY(load_A(A, H_, ar, 0, lane), load_A(A, H_, ar, k0, lane))
  v8f accs[4] = {acc0, acc1, acc2, acc3};
  int m0 = tm * 16 + ((lane >> 4) << 3);
#pragma unroll
  for (int j = 0; j < 4; ++j) {
    int n = tg * 64 + j * 16 + (lane & 15);
    float bz = bias[n];
#pragma unroll
    for (int r = 0; r < 8; ++r) {
      int b = m0 + r;
      float v = accs[j][r] + bz;
      bf16 o = isidT[(size_t)b * S_] ? (bf16)v : embt[(size_t)b * S_ * E_ + n];
      X[(size_t)b * H_ + n] = o;
    }
  }
}

// ---------------------------------------------------------------------------
// LSTM gates: C = A1@B1^T + A2@B2^T + bias.  M=32, N=2048 (Ngrp=32), K=2*512.
// Single pipelined K=1024 loop, sources switch at the 512 boundary.
// ---------------------------------------------------------------------------
__global__ void k_gemm_dual4(const bf16* __restrict__ A1, const bf16* __restrict__ B1,
                             const bf16* __restrict__ A2, const bf16* __restrict__ B2,
                             const float* __restrict__ bias,
                             float* __restrict__ Cf, int Ngrp) {
  int wid  = (blockIdx.x * blockDim.x + threadIdx.x) >> 5;
  int lane = threadIdx.x & 31;
  int tm = wid / Ngrp, tg = wid - tm * Ngrp;
  if (tm * 16 >= B_) return;
  int ar = tm * 16 + (lane & 15);
  int nb = tg * 64 + (lane & 15);
  const int K = 2 * H_, ldb = H_;
#define LD_A2(K0) ((K0) < H_ ? load_A(A1, H_, ar, (K0), lane)                  \
                             : load_A(A2, H_, ar, (K0) - H_, lane))
#define LD_B2(NR, K0) ((K0) < H_ ? load_B(B1, ldb, (NR), (K0), lane)           \
                                 : load_B(B2, ldb, (NR), (K0) - H_, lane))
  v8f acc0 = ACC_ZERO, acc1 = ACC_ZERO, acc2 = ACC_ZERO, acc3 = ACC_ZERO;
  v16bf a  = LD_A2(0);
  v16bf b0 = LD_B2(nb, 0), b1 = LD_B2(nb + 16, 0);
  v16bf b2 = LD_B2(nb + 32, 0), b3 = LD_B2(nb + 48, 0);
  for (int k0 = 32; k0 < K; k0 += 32) {
    v16bf an = LD_A2(k0);
    v16bf c0 = LD_B2(nb, k0), c1 = LD_B2(nb + 16, k0);
    v16bf c2 = LD_B2(nb + 32, k0), c3 = LD_B2(nb + 48, k0);
    acc0 = wmma_bf16(a, b0, acc0);
    acc1 = wmma_bf16(a, b1, acc1);
    acc2 = wmma_bf16(a, b2, acc2);
    acc3 = wmma_bf16(a, b3, acc3);
    a = an; b0 = c0; b1 = c1; b2 = c2; b3 = c3;
  }
  acc0 = wmma_bf16(a, b0, acc0);
  acc1 = wmma_bf16(a, b1, acc1);
  acc2 = wmma_bf16(a, b2, acc2);
  acc3 = wmma_bf16(a, b3, acc3);
#undef LD_A2
#undef LD_B2
  v8f accs[4] = {acc0, acc1, acc2, acc3};
  int m0 = tm * 16 + ((lane >> 4) << 3);
#pragma unroll
  for (int j = 0; j < 4; ++j) {
    int n = tg * 64 + j * 16 + (lane & 15);
    float bz = bias[n];
#pragma unroll
    for (int r = 0; r < 8; ++r)
      Cf[(size_t)(m0 + r) * (4 * H_) + n] = accs[j][r] + bz;
  }
}

// ---------------------------------------------------------------------------
// Stack update: val = tanh(sup @ uv_Wt + uv_b); new = old*(1-g)+g*val; mask->0.
// M = B*STACK = 1024, K = 1024 (stack|o), N = 512 (Ngrp = 8). 16x64/wave to
// keep 512 waves in flight (occupancy > intensity for this latency-bound step).
// ---------------------------------------------------------------------------
__global__ void k_gemm_stack4(const bf16* __restrict__ stB, const bf16* __restrict__ ob,
                              const bf16* __restrict__ Bt, const float* __restrict__ uvb,
                              const float* __restrict__ gate,
                              const float* __restrict__ stOldF,
                              const unsigned char* __restrict__ updT, // + t*STACK_
                              float* __restrict__ stNewF, bf16* __restrict__ stNewB,
                              int Ngrp) {
  int wid  = (blockIdx.x * blockDim.x + threadIdx.x) >> 5;
  int lane = threadIdx.x & 31;
  int tm = wid / Ngrp, tg = wid - tm * Ngrp;
  if (tm * 16 >= B_ * STACK_) return;
  int ar = tm * 16 + (lane & 15);
  int nb = tg * 64 + (lane & 15);
  const int K = 2 * H_, ldb = 2 * H_;
  GEMM4_BODY(load_A_split(stB, ob, ar, 0, lane),
             load_A_split(stB, ob, ar, k0, lane))
  v8f accs[4] = {acc0, acc1, acc2, acc3};
  int m0 = tm * 16 + ((lane >> 4) << 3);
#pragma unroll
  for (int j = 0; j < 4; ++j) {
    int n = tg * 64 + j * 16 + (lane & 15);
    float bz = uvb[n];
#pragma unroll
    for (int r = 0; r < 8; ++r) {
      int m = m0 + r;
      int b = m >> 5, s = m & (STACK_ - 1);
      float g   = gate[m];
      float val = tanhf(accs[j][r] + bz);
      float old = stOldF[(size_t)m * H_ + n];
      float nv  = old * (1.f - g) + g * val;
      if (updT[(size_t)b * S_ * STACK_ + s]) nv = 0.f;
      stNewF[(size_t)m * H_ + n] = nv;
      stNewB[(size_t)m * H_ + n] = (bf16)nv;
    }
  }
}

// ------------------------- elementwise / prep kernels -----------------------

__global__ void k_embed(const int* __restrict__ tokens, const float* __restrict__ table,
                        bf16* __restrict__ emb) {
  size_t tid = (size_t)blockIdx.x * 256 + threadIdx.x;   // B*S*E threads
  int e = (int)(tid & (E_ - 1));
  size_t bs = tid >> 9;
  int tok = tokens[bs];
  emb[tid] = (bf16)table[(size_t)tok * E_ + e];
}

__global__ void k_cell(const float* __restrict__ gates, float* __restrict__ c,
                       bf16* __restrict__ hB, bf16* __restrict__ featT) {
  int tid = blockIdx.x * 256 + threadIdx.x;              // B*H threads
  int b = tid >> 9, j = tid & (H_ - 1);
  const float* g = gates + (size_t)b * 4 * H_;
  float i  = sigm(g[j]);
  float f  = sigm(g[H_ + j]);
  float gg = tanhf(g[2 * H_ + j]);
  float o  = sigm(g[3 * H_ + j]);
  float cn = f * c[tid] + i * gg;
  c[tid] = cn;
  float h = o * tanhf(cn);
  hB[tid] = (bf16)h;
  if (featT) featT[(size_t)b * S_ * H_ + j] = (bf16)h;
}

__global__ void k_gate(const bf16* __restrict__ stB, const bf16* __restrict__ ob,
                       const float* __restrict__ ugW, const float* __restrict__ ugb,
                       float* __restrict__ gate) {
  int row = blockIdx.x * 256 + threadIdx.x;              // B*STACK threads
  if (row >= B_ * STACK_) return;
  const bf16* sr = stB + (size_t)row * H_;
  const bf16* orow = ob + (size_t)(row >> 5) * H_;
  float a = ugb[0];
  for (int j = 0; j < H_; ++j)
    a += (float)sr[j] * ugW[j] + (float)orow[j] * ugW[H_ + j];
  gate[row] = sigm(a);
}

// dst[Cpad, R]: dst[c*R + r] = (c < C) ? src[r*C + c] : 0   (fp32 -> bf16)
__global__ void k_transpose(const float* __restrict__ src, bf16* __restrict__ dst,
                            int R, int C, int Cpad) {
  int tid = blockIdx.x * 256 + threadIdx.x;
  if (tid >= Cpad * R) return;
  int c = tid / R, r = tid - c * R;
  dst[tid] = (c < C) ? (bf16)src[(size_t)r * C + c] : (bf16)0.f;
}

__global__ void k_convert(const float* __restrict__ src, bf16* __restrict__ dst, int n) {
  int tid = blockIdx.x * 256 + threadIdx.x;
  if (tid < n) dst[tid] = (bf16)src[tid];
}

__global__ void k_addbias(const float* __restrict__ a, const float* __restrict__ b,
                          float* __restrict__ dst, int n) {
  int tid = blockIdx.x * 256 + threadIdx.x;
  if (tid < n) dst[tid] = a[tid] + b[tid];
}

__global__ void k_init(float* __restrict__ stF0, bf16* __restrict__ stB0,
                       const float* __restrict__ h0in, const float* __restrict__ c0in,
                       bf16* __restrict__ h0b, bf16* __restrict__ h1b,
                       float* __restrict__ c0l, float* __restrict__ c1l) {
  int tid = blockIdx.x * 256 + threadIdx.x;              // B*STACK*H threads
  if (tid < B_ * STACK_ * H_) { stF0[tid] = 0.f; stB0[tid] = (bf16)0.f; }
  if (tid < B_ * H_) {
    h0b[tid] = (bf16)h0in[tid];
    h1b[tid] = (bf16)h0in[B_ * H_ + tid];
    c0l[tid] = c0in[tid];
    c1l[tid] = c0in[B_ * H_ + tid];
  }
}

// ============================================================================
extern "C" void kernel_launch(void* const* d_in, const int* in_sizes, int n_in,
                              void* d_out, int out_size, void* d_ws, size_t ws_size,
                              hipStream_t stream) {
  (void)in_sizes; (void)n_in; (void)out_size; (void)ws_size;

  const int*           tokens = (const int*)d_in[0];
  const int*           sidx   = (const int*)d_in[1];
  const unsigned char* isid   = (const unsigned char*)d_in[2];
  const unsigned char* updm   = (const unsigned char*)d_in[3];
  const unsigned char* tmask  = (const unsigned char*)d_in[4];
  const unsigned char* hid    = (const unsigned char*)d_in[5];
  const float* etab = (const float*)d_in[6];
  const float* h0in = (const float*)d_in[7];
  const float* c0in = (const float*)d_in[8];
  const float* Wih0 = (const float*)d_in[9];
  const float* Whh0 = (const float*)d_in[10];
  const float* bih0 = (const float*)d_in[11];
  const float* bhh0 = (const float*)d_in[12];
  const float* Wih1 = (const float*)d_in[13];
  const float* Whh1 = (const float*)d_in[14];
  const float* bih1 = (const float*)d_in[15];
  const float* bhh1 = (const float*)d_in[16];
  const float* stW1 = (const float*)d_in[17];
  const float* stb1 = (const float*)d_in[18];
  const float* stW2 = (const float*)d_in[19];
  const float* stb2 = (const float*)d_in[20];
  const float* ugW  = (const float*)d_in[21];
  const float* ugb  = (const float*)d_in[22];
  const float* uvW  = (const float*)d_in[23];
  const float* uvb  = (const float*)d_in[24];
  const float* kwW1 = (const float*)d_in[25];
  const float* kwb1 = (const float*)d_in[26];
  const float* kwW2 = (const float*)d_in[27];
  const float* kwb2 = (const float*)d_in[28];
  const float* oW1  = (const float*)d_in[29];
  const float* ob1  = (const float*)d_in[30];
  const float* oW2  = (const float*)d_in[31];
  const float* ob2  = (const float*)d_in[32];
  float* out = (float*)d_out;

  char* wsb = (char*)d_ws;
  size_t off = 0;
  auto alloc = [&](size_t bytes) -> void* {
    void* p = wsb + off;
    off += (bytes + 255) & ~(size_t)255;
    return p;
  };

  bf16* embB  = (bf16*)alloc((size_t)B_ * S_ * E_ * 2);
  bf16* featB = (bf16*)alloc((size_t)B_ * S_ * H_ * 2);
  bf16* tmpB  = (bf16*)alloc((size_t)B_ * S_ * H_ * 2);
  float* stF[2]; bf16* stBb[2];
  stF[0]  = (float*)alloc((size_t)B_ * STACK_ * H_ * 4);
  stF[1]  = (float*)alloc((size_t)B_ * STACK_ * H_ * 4);
  stBb[0] = (bf16*)alloc((size_t)B_ * STACK_ * H_ * 2);
  stBb[1] = (bf16*)alloc((size_t)B_ * STACK_ * H_ * 2);
  bf16* W_st1 = (bf16*)alloc((size_t)512 * 1024 * 2);
  bf16* W_st2 = (bf16*)alloc((size_t)512 * 512 * 2);
  bf16* W_ih0 = (bf16*)alloc((size_t)2048 * 512 * 2);
  bf16* W_hh0 = (bf16*)alloc((size_t)2048 * 512 * 2);
  bf16* W_ih1 = (bf16*)alloc((size_t)2048 * 512 * 2);
  bf16* W_hh1 = (bf16*)alloc((size_t)2048 * 512 * 2);
  bf16* W_uv  = (bf16*)alloc((size_t)512 * 1024 * 2);
  bf16* W_kw1 = (bf16*)alloc((size_t)512 * 512 * 2);
  bf16* W_kw2 = (bf16*)alloc((size_t)KWPAD_ * 512 * 2);
  bf16* W_o1  = (bf16*)alloc((size_t)512 * 512 * 2);
  bf16* W_o2  = (bf16*)alloc((size_t)IDPAD_ * 512 * 2);
  float* bias0 = (float*)alloc(2048 * 4);
  float* bias1 = (float*)alloc(2048 * 4);
  bf16* t1   = (bf16*)alloc((size_t)B_ * H_ * 2);
  bf16* xB   = (bf16*)alloc((size_t)B_ * H_ * 2);
  float* gates = (float*)alloc((size_t)B_ * 2048 * 4);
  bf16* h0b = (bf16*)alloc((size_t)B_ * H_ * 2);
  bf16* h1b = (bf16*)alloc((size_t)B_ * H_ * 2);
  float* c0l = (float*)alloc((size_t)B_ * H_ * 4);
  float* c1l = (float*)alloc((size_t)B_ * H_ * 4);
  float* gbuf = (float*)alloc((size_t)B_ * STACK_ * 4);

  dim3 blk(256);

  // ---- one-time weight prep (re-done every call: deterministic) ----
  k_transpose<<<(512 * 1024) / 256, blk, 0, stream>>>(stW1, W_st1, 1024, 512, 512);
  k_transpose<<<(512 * 512) / 256, blk, 0, stream>>>(stW2, W_st2, 512, 512, 512);
  k_transpose<<<(512 * 1024) / 256, blk, 0, stream>>>(uvW, W_uv, 1024, 512, 512);
  k_transpose<<<(512 * 512) / 256, blk, 0, stream>>>(kwW1, W_kw1, 512, 512, 512);
  k_transpose<<<(KWPAD_ * 512) / 256, blk, 0, stream>>>(kwW2, W_kw2, 512, KW_, KWPAD_);
  k_transpose<<<(512 * 512) / 256, blk, 0, stream>>>(oW1, W_o1, 512, 512, 512);
  k_transpose<<<(IDPAD_ * 512) / 256, blk, 0, stream>>>(oW2, W_o2, 512, IDV_, IDPAD_);
  k_convert<<<(2048 * 512) / 256, blk, 0, stream>>>(Wih0, W_ih0, 2048 * 512);
  k_convert<<<(2048 * 512) / 256, blk, 0, stream>>>(Whh0, W_hh0, 2048 * 512);
  k_convert<<<(2048 * 512) / 256, blk, 0, stream>>>(Wih1, W_ih1, 2048 * 512);
  k_convert<<<(2048 * 512) / 256, blk, 0, stream>>>(Whh1, W_hh1, 2048 * 512);
  k_addbias<<<8, blk, 0, stream>>>(bih0, bhh0, bias0, 2048);
  k_addbias<<<8, blk, 0, stream>>>(bih1, bhh1, bias1, 2048);
  k_embed<<<(B_ * S_ * E_) / 256, blk, 0, stream>>>(tokens, etab, embB);
  k_init<<<(B_ * STACK_ * H_) / 256, blk, 0, stream>>>(stF[0], stBb[0], h0in, c0in,
                                                       h0b, h1b, c0l, c1l);

  // ---- recurrent scan over S steps ----
  for (int t = 0; t < S_; ++t) {
    int p = t & 1;
    const bf16* embt = embB + (size_t)t * E_;

    // tmp1 = relu([stack[b,sidx] | emb_t] @ st_W1 + st_b1)   (hcat fused)
    k_gemm_hcat4<<<2, blk, 0, stream>>>(stBb[p], embt, sidx + t, W_st1, 1024,
                                        stb1, t1, 8, 1024);
    // x = isid ? (tmp1 @ st_W2 + st_b2) : emb_t
    k_gemm_scoped4<<<2, blk, 0, stream>>>(t1, W_st2, 512, stb2, embt, isid + t,
                                          xB, 8, 512);
    // layer-0 gates + cell
    k_gemm_dual4<<<8, blk, 0, stream>>>(xB, W_ih0, h0b, W_hh0, bias0, gates, 32);
    k_cell<<<(B_ * H_) / 256, blk, 0, stream>>>(gates, c0l, h0b, nullptr);
    // layer-1 gates + cell (o = h1, also stored to feat[:, t, :])
    k_gemm_dual4<<<8, blk, 0, stream>>>(h0b, W_ih1, h1b, W_hh1, bias1, gates, 32);
    k_cell<<<(B_ * H_) / 256, blk, 0, stream>>>(gates, c1l, h1b,
                                                featB + (size_t)t * H_);
    // stack update (ping-pong)
    k_gate<<<(B_ * STACK_) / 256, blk, 0, stream>>>(stBb[p], h1b, ugW, ugb, gbuf);
    k_gemm_stack4<<<64, blk, 0, stream>>>(stBb[p], h1b, W_uv, uvb, gbuf, stF[p],
                                          updm + (size_t)t * STACK_,
                                          stF[1 - p], stBb[1 - p], 8);
  }

  // ---- output heads (32x64-per-wave GEMMs) ----
  // keyword head: waves = (8192/32) * 8 = 2048 -> 256 blocks
  k_gemm8<<<256, blk, 0, stream>>>(featB, H_, W_kw1, H_, kwb1, tmpB, H_,
                                   B_ * S_, 8, H_);
  // waves = 256 * 16 = 4096 -> 512 blocks
  k_gemm_kw8<<<512, blk, 0, stream>>>(tmpB, W_kw2, kwb2, tmask, out, KWPAD_ / 64);
  // identifier head
  k_gemm8<<<256, blk, 0, stream>>>(featB, H_, W_o1, H_, ob1, tmpB, H_,
                                   B_ * S_, 8, H_);
  // waves = 256 * 141 = 36096 -> 4512 blocks
  k_gemm_ident8<<<4512, blk, 0, stream>>>(tmpB, W_o2, ob2, hid, out, IDPAD_ / 64);
}
